// BNBLinearINT8_39204461478926
// MI455X (gfx1250) — 8-bit matrix operand order repaired and checked by reading, compiled, not run on hardware
//
#include <hip/hip_runtime.h>
#include <stdint.h>


typedef __attribute__((ext_vector_type(8))) int v8i;

static constexpr int M_ = 8192;
static constexpr int N_ = 11008;
static constexpr int K_ = 4096;

#define THRESH 6.0f
#define EPSQ   1e-8f

__global__ void k_init(int* __restrict__ cnt) {
  if (threadIdx.x == 0 && blockIdx.x == 0) *cnt = 0;
}

__global__ void k_colmask(const float* __restrict__ x, int* __restrict__ mask,
                          int* __restrict__ list, int* __restrict__ cnt) {
  int j = blockIdx.x * blockDim.x + threadIdx.x;
  if (j >= K_) return;
  int m = 0;
  for (int i = 0; i < M_; ++i)
    m |= (fabsf(x[(size_t)i * K_ + j]) > THRESH) ? 1 : 0;
  mask[j] = m;
  if (m) {
    int p = atomicAdd(cnt, 1);
    list[p] = j;
  }
}

__global__ void k_quantW(const float* __restrict__ W, int8_t* __restrict__ wq,
                         float* __restrict__ sw) {
  const int o   = blockIdx.x;
  const int tid = threadIdx.x;
  const float* row = W + (size_t)o * K_;

  float m = 0.f;
  for (int j = tid; j < K_; j += 256) m = fmaxf(m, fabsf(row[j]));

  __shared__ float red[256];
  red[tid] = m;
  __syncthreads();
  for (int s = 128; s > 0; s >>= 1) {
    if (tid < s) red[tid] = fmaxf(red[tid], red[tid + s]);
    __syncthreads();
  }
  const float sc = red[0] / 127.f + EPSQ;
  if (tid == 0) sw[o] = sc;

  for (int j = tid; j < K_; j += 256) {
    float q = rintf(row[j] / sc);
    q = fminf(127.f, fmaxf(-127.f, q));
    wq[(size_t)o * K_ + j] = (int8_t)(int)q;
  }
}

__global__ void k_quantX(const float* __restrict__ x, const int* __restrict__ mask,
                         int8_t* __restrict__ xq, float* __restrict__ sx) {
  const int i   = blockIdx.x;
  const int tid = threadIdx.x;
  const float* row = x + (size_t)i * K_;

  float m = 0.f;
  for (int j = tid; j < K_; j += 256) {
    float v = mask[j] ? 0.f : row[j];
    m = fmaxf(m, fabsf(v));
  }
  __shared__ float red[256];
  red[tid] = m;
  __syncthreads();
  for (int s = 128; s > 0; s >>= 1) {
    if (tid < s) red[tid] = fmaxf(red[tid], red[tid + s]);
    __syncthreads();
  }
  const float sc = red[0] / 127.f + EPSQ;
  if (tid == 0) sx[i] = sc;

  for (int j = tid; j < K_; j += 256) {
    float v = mask[j] ? 0.f : row[j];
    float q = rintf(v / sc);
    q = fminf(127.f, fmaxf(-127.f, q));
    xq[(size_t)i * K_ + j] = (int8_t)(int)q;
  }
}

__global__ __launch_bounds__(256) void k_gemm(
    const int8_t* __restrict__ xq, const int8_t* __restrict__ wq,
    const float* __restrict__ sx, const float* __restrict__ sw,
    const float* __restrict__ bias, float* __restrict__ out) {
  constexpr int ROWB = 80;
  constexpr int BUFB = 128 * ROWB;
  __shared__ uint8_t smA[2 * BUFB];
  __shared__ uint8_t smB[2 * BUFB];

  const int tid  = threadIdx.x;
  const int lane = tid & 31;
  const int wave = tid >> 5;
  const int wm   = wave >> 2;
  const int wn   = wave & 3;
  const int bm   = blockIdx.y * 128;
  const int bn   = blockIdx.x * 128;

  const int l15    = lane & 15;
  const int half8  = (lane < 16) ? 0 : 8;
  const int half16 = (lane < 16) ? 0 : 16;

  const int srow = tid >> 1;
  const int sseg = (tid & 1) * 32;

  const uint32_t ldsA = (uint32_t)(uintptr_t)(&smA[0]) + srow * ROWB + sseg;
  const uint32_t ldsB = (uint32_t)(uintptr_t)(&smB[0]) + srow * ROWB + sseg;
  const uint64_t gA   = (uint64_t)(uintptr_t)(xq + (size_t)(bm + srow) * K_ + sseg);
  const uint64_t gB   = (uint64_t)(uintptr_t)(wq + (size_t)(bn + srow) * K_ + sseg);

  auto issue = [&](int buf, int kk) {
    const uint32_t la = ldsA + buf * BUFB;
    const uint32_t lb = ldsB + buf * BUFB;
    const uint64_t ga = gA + kk;
    const uint64_t gb = gB + kk;
    asm volatile("global_load_async_to_lds_b128 %0, %1, off"
                 :: "v"(la), "v"(ga) : "memory");
    asm volatile("global_load_async_to_lds_b128 %0, %1, off offset:16"
                 :: "v"(la), "v"(ga) : "memory");
    asm volatile("global_load_async_to_lds_b128 %0, %1, off"
                 :: "v"(lb), "v"(gb) : "memory");
    asm volatile("global_load_async_to_lds_b128 %0, %1, off offset:16"
                 :: "v"(lb), "v"(gb) : "memory");
  };

  v8i acc[4][2] = {};

  auto compute = [&](int buf) {
    const uint8_t* baseA = &smA[buf * BUFB];
    const uint8_t* baseB = &smB[buf * BUFB];
    v8i bf[2];
    #pragma unroll
    for (int nt = 0; nt < 2; ++nt) {
      const uint8_t* pb = baseB + (size_t)(wn * 32 + nt * 16 + l15) * ROWB + half16;
      const uint4 p0 = *(const uint4*)(pb);
      const uint4 p1 = *(const uint4*)(pb + 32);
      v8i t = { (int)p0.x, (int)p0.y, (int)p0.z, (int)p0.w,
                (int)p1.x, (int)p1.y, (int)p1.z, (int)p1.w };
      bf[nt] = t;
    }
    #pragma unroll
    for (int mt = 0; mt < 4; ++mt) {
      const uint8_t* pa = baseA + (size_t)(wm * 64 + mt * 16 + l15) * ROWB + half16;
      const uint2 q0 = *(const uint2*)(pa);
      const uint2 q1 = *(const uint2*)(pa + 8);
      const uint2 q2 = *(const uint2*)(pa + 32);
      const uint2 q3 = *(const uint2*)(pa + 40);
      v8i af = { (int)q0.x, (int)q0.y, (int)q1.x, (int)q1.y,
                 (int)q2.x, (int)q2.y, (int)q3.x, (int)q3.y };
      acc[mt][0] = __builtin_amdgcn_wmma_i32_16x16x64_iu8(
          true, af, true, bf[0], acc[mt][0], false, false);
      acc[mt][1] = __builtin_amdgcn_wmma_i32_16x16x64_iu8(
          true, af, true, bf[1], acc[mt][1], false, false);
    }
  };

  constexpr int NS = K_ / 64;
  issue(0, 0);
  issue(1, 64);
  int buf = 0;
  for (int s = 0; s < NS; ++s) {
    if (s + 1 < NS) asm volatile("s_wait_asynccnt 0x4" ::: "memory");
    else            asm volatile("s_wait_asynccnt 0x0" ::: "memory");
    __syncthreads();
    compute(buf);
    __syncthreads();
    if (s + 2 < NS) issue(buf, (s + 2) * 64);
    buf ^= 1;
  }

  #pragma unroll
  for (int nt = 0; nt < 2; ++nt) {
    const int gc  = bn + wn * 32 + nt * 16 + l15;
    const float swv = sw[gc];
    const float bv  = bias[gc];
    #pragma unroll
    for (int mt = 0; mt < 4; ++mt) {
      const int mbase = bm + wm * 64 + mt * 16 + half8;
      #pragma unroll
      for (int r = 0; r < 8; ++r) {
        const int gr = mbase + r;
        out[(size_t)gr * N_ + gc] = (float)acc[mt][nt][r] * sx[gr] * swv + bv;
      }
    }
  }
}

__global__ void k_outfix(const float* __restrict__ x, const int8_t* __restrict__ wq,
                         const float* __restrict__ sw, const int* __restrict__ list,
                         const int* __restrict__ cnt, float* __restrict__ out) {
  const int nc = *cnt;
  if (nc == 0) return;
  const int tid  = threadIdx.x;
  const int row  = blockIdx.y * 128 + (tid >> 1);
  const int cb   = blockIdx.x * 128 + (tid & 1) * 64;
  for (int t = 0; t < nc; ++t) {
    const int c = list[t];
    const float xv = x[(size_t)row * K_ + c];
    if (xv == 0.f) continue;
    for (int o = 0; o < 64; ++o) {
      const int gc = cb + o;
      out[(size_t)row * N_ + gc] += xv * (float)wq[(size_t)gc * K_ + c] * sw[gc];
    }
  }
}

extern "C" void kernel_launch(void* const* d_in, const int* in_sizes, int n_in,
                              void* d_out, int out_size, void* d_ws, size_t ws_size,
                              hipStream_t stream) {
  (void)in_sizes; (void)n_in; (void)out_size; (void)ws_size;

  const float* x = (const float*)d_in[0];
  const float* W = (const float*)d_in[1];
  const float* b = (const float*)d_in[2];
  float* out = (float*)d_out;

  uint8_t* ws = (uint8_t*)d_ws;
  int8_t* wq   = (int8_t*)ws;
  int8_t* xq   = wq + (size_t)N_ * K_;
  float*  sw   = (float*)(xq + (size_t)M_ * K_);
  float*  sx   = sw + N_;
  int*    mask = (int*)(sx + M_);
  int*    list = mask + K_;
  int*    cnt  = list + K_;

  k_init<<<1, 64, 0, stream>>>(cnt);
  k_colmask<<<K_ / 256, 256, 0, stream>>>(x, mask, list, cnt);
  k_quantW<<<N_, 256, 0, stream>>>(W, wq, sw);
  k_quantX<<<M_, 256, 0, stream>>>(x, mask, xq, sx);

  dim3 grid(N_ / 128, M_ / 128);
  k_gemm<<<grid, 256, 0, stream>>>(xq, wq, sx, sw, b, out);
  k_outfix<<<grid, 256, 0, stream>>>(x, wq, sw, list, cnt, out);
}
